// AgnosticInteractionBlock_21406117003825
// MI455X (gfx1250) — matrix operand order repaired and checked by reading, not compiled, not run on hardware
//
#include <hip/hip_runtime.h>
#include <hip/hip_bf16.h>

// ---------------------------------------------------------------------------
// MACE-style interaction block for MI455X (gfx1250, wave32, WMMA).
// - All GEMMs on v_wmma_f32_16x16x32_bf16 (bf16 in, f32 accumulate).
// - Weights pre-swizzled into WMMA B-fragment order: 2x global_load_b128/frag.
// - B fragments batched + double-buffered so wmma chains don't stall on
//   s_wait_loadcnt 0x0 (weight loads overlap the previous tile's WMMAs).
// - A fragments hoisted into VGPRs (loaded once per GEMM from LDS).
// - Tile staging via global_load_async_to_lds_b128 (ASYNCcnt path).
// - Edge gather prefetched (global_prefetch_b8) to overlap the radial MLP.
// - Fused edge MLP + tensor-product messages + scatter: tpw (E x 512) never
//   touches HBM; m0/m1 accumulators (82 MB) are L2-resident for the atomics.
// ---------------------------------------------------------------------------

#define N_NODES 20000
#define N_EDGES 160000
#define N_ELEM  10

typedef __bf16 bf16_t;
typedef __attribute__((ext_vector_type(16))) __bf16 v16bf;
typedef __attribute__((ext_vector_type(8)))  float  v8f;

__device__ __forceinline__ v8f wmma_bf16(v16bf a, v16bf b, v8f c) {
  // 8 args: (neg_a, A, neg_b, B, c_mod, C, reuse_a, reuse_b)
  return __builtin_amdgcn_wmma_f32_16x16x32_bf16(false, a, false, b, (short)0,
                                                 c, false, false);
}

// A fragment: 16x32 bf16, element (m,k) = src[m*row_stride + k*col_stride].
// Layout used here (contiguous k, B packed the same): lane m=lane&15, group g=lane>>4;
// a[idx] holds K = idx + 16*g.
__device__ __forceinline__ v16bf load_a_frag(const float* __restrict__ src,
                                             int row_stride, int col_stride) {
  const int lane = threadIdx.x & 31;
  const int m = lane & 15;
  const int g = lane >> 4;
  const float* row = src + m * row_stride;
  v16bf a;
#pragma unroll
  for (int idx = 0; idx < 16; ++idx) {
    const int k = idx + 16 * g;
    a[idx] = (__bf16)row[k * col_stride];
  }
  return a;
}

// Packed B fragment: each 32x16 tile stored as [lane][16] contiguous bf16,
// so a fragment is a single 32-byte aligned per-lane vector load.
__device__ __forceinline__ v16bf load_b_packed(const bf16_t* __restrict__ base,
                                               int tile) {
  const int lane = threadIdx.x & 31;
  return *(const v16bf*)(base + ((size_t)tile << 9) + (lane << 4));
}

// Batched B fragments for a K-chain (tile indices kt*ntn + nt).
template <int KT>
struct BFrags {
  v16bf b[KT];
};

template <int KT>
__device__ __forceinline__ void load_bfrags(BFrags<KT>& f,
                                            const bf16_t* __restrict__ base,
                                            int nt, int ntn) {
#pragma unroll
  for (int kt = 0; kt < KT; ++kt) f.b[kt] = load_b_packed(base, kt * ntn + nt);
}

template <int KT>
__device__ __forceinline__ v8f wmma_chain(const v16bf* a, const BFrags<KT>& f) {
  v8f c = {};
#pragma unroll
  for (int kt = 0; kt < KT; ++kt) c = wmma_bf16(a[kt], f.b[kt], c);
  return c;
}

__device__ __forceinline__ float silu(float x) {
  return x / (1.0f + __expf(-x));
}

// Async global->LDS 16B copy (VFLAT async op, tracked by ASYNCcnt).
__device__ __forceinline__ void async_copy_b128(void* lds_ptr,
                                                const void* gaddr) {
  unsigned lds_off = (unsigned)(uintptr_t)lds_ptr;  // low 32 bits = LDS offset
  unsigned long long ga = (unsigned long long)(uintptr_t)gaddr;
  asm volatile("global_load_async_to_lds_b128 %0, %1, off"
               :
               : "v"(lds_off), "v"(ga)
               : "memory");
}
__device__ __forceinline__ void wait_async0() {
  asm volatile("s_wait_asynccnt 0x0" ::: "memory");
}

// ---------------------------------------------------------------------------
// Weight pre-swizzle into WMMA B-fragment order.
// dst[tile*512 + lane*16 + idx] = (bf16) src[k*ld + n]
//   tile = kt*(N/16)+nt, k = kt*32 + idx + 16*(lane>>4), n = nt*16 + (lane&15)
// ---------------------------------------------------------------------------
__global__ void k_pack_b(const float* __restrict__ src, bf16_t* __restrict__ dst,
                         int K, int N, int ld) {
  int i = blockIdx.x * blockDim.x + threadIdx.x;
  if (i >= K * N) return;
  int tile = i >> 9, rem = i & 511;
  int lane = rem >> 4, idx = rem & 15;
  int ntn = N >> 4;
  int kt = tile / ntn, nt = tile - kt * ntn;
  int k = kt * 32 + idx + 16 * (lane >> 4);
  int n = nt * 16 + (lane & 15);
  dst[i] = (__bf16)src[(size_t)k * ld + n];
}

// W_sk (128, 10, 128): pack 10 per-element B matrices B_v[u][k] = src[u*1280+v*128+k]
__global__ void k_pack_sk(const float* __restrict__ src,
                          bf16_t* __restrict__ dst) {
  int i = blockIdx.x * blockDim.x + threadIdx.x;
  if (i >= N_ELEM * 16384) return;
  int v = i >> 14, j = i & 16383;
  int tile = j >> 9, rem = j & 511;
  int lane = rem >> 4, idx = rem & 15;
  int kt = tile >> 3, nt = tile & 7;  // K=128 -> 4 ktiles, N=128 -> 8 ntiles
  int k = kt * 32 + idx + 16 * (lane >> 4);
  int n = nt * 16 + (lane & 15);
  dst[i] = (__bf16)src[(size_t)k * 1280 + v * 128 + n];
}

__global__ void k_zero(float* __restrict__ p, int n) {
  int i = blockIdx.x * blockDim.x + threadIdx.x;
  if (i < n) p[i] = 0.0f;
}

// ---------------------------------------------------------------------------
// Node up-projection: x0u = x0 @ Wup0 * inv ; x1u[n,k,m] = sum_u x1[n,u,m]Wup1[u,k]*inv
// One wave per 16-node tile; 16x512 tile async-staged to LDS; 128 WMMA/tile.
// ---------------------------------------------------------------------------
__global__ __launch_bounds__(32) void k_node_up(
    const float* __restrict__ node_feats,
    const bf16_t* __restrict__ Wup0, const bf16_t* __restrict__ Wup1,
    float* __restrict__ x0u, float* __restrict__ x1u) {
  __shared__ float lds[16 * 512];
  const int n0 = blockIdx.x * 16;
  const int lane = threadIdx.x;
  const int col = lane & 15, g = lane >> 4;
  const float inv = 0.08838834764831845f;  // 1/sqrt(128)

  {
    const float* gsrc = node_feats + (size_t)n0 * 512;
    for (int i = lane * 4; i < 16 * 512; i += 32 * 4)
      async_copy_b128(&lds[i], gsrc + i);
    wait_async0();
  }
  __syncthreads();

  // x0 @ Wup0 (K=128): hoist 4 A-frags, double-buffer B across 8 N-tiles
  {
    v16bf a0[4];
#pragma unroll
    for (int kt = 0; kt < 4; ++kt) a0[kt] = load_a_frag(lds + kt * 32, 512, 1);
    BFrags<4> cb, nb;
    load_bfrags<4>(cb, Wup0, 0, 8);
#pragma unroll
    for (int nt = 0; nt < 8; ++nt) {
      if (nt < 7) load_bfrags<4>(nb, Wup0, nt + 1, 8);
      v8f c = wmma_chain<4>(a0, cb);
#pragma unroll
      for (int r = 0; r < 8; ++r) {
        int row = r + 8 * g;
        x0u[(size_t)(n0 + row) * 128 + nt * 16 + col] = c[r] * inv;
      }
      cb = nb;
    }
  }

  // x1 einsum per m-component: A element (row,k) = lds[row*512 + 128 + k*3 + mc]
  for (int mc = 0; mc < 3; ++mc) {
    v16bf a1[4];
#pragma unroll
    for (int kt = 0; kt < 4; ++kt)
      a1[kt] = load_a_frag(lds + 128 + mc + kt * 96, 512, 3);
    BFrags<4> cb, nb;
    load_bfrags<4>(cb, Wup1, 0, 8);
#pragma unroll
    for (int nt = 0; nt < 8; ++nt) {
      if (nt < 7) load_bfrags<4>(nb, Wup1, nt + 1, 8);
      v8f c = wmma_chain<4>(a1, cb);
#pragma unroll
      for (int r = 0; r < 8; ++r) {
        int row = r + 8 * g;
        x1u[(size_t)(n0 + row) * 384 + (nt * 16 + col) * 3 + mc] = c[r] * inv;
      }
      cb = nb;
    }
  }
}

// ---------------------------------------------------------------------------
// Fused edge pipeline: prefetch gather -> radial MLP (layer1 VALU, 2-4 WMMA)
// -> tensor-product messages -> scatter-add (f32 atomics, L2-resident m0/m1).
// ---------------------------------------------------------------------------
__global__ __launch_bounds__(32) void k_edge(
    const float* __restrict__ edge_attrs, const float* __restrict__ edge_feats,
    const int* __restrict__ edge_index, const float* __restrict__ W_fc1,
    const bf16_t* __restrict__ Wfc2, const bf16_t* __restrict__ Wfc3,
    const bf16_t* __restrict__ Wfc4,
    const float* __restrict__ x0u, const float* __restrict__ x1u,
    float* __restrict__ m0, float* __restrict__ m1) {
  __shared__ float ef[16 * 8];
  __shared__ float wfc1[8 * 64];
  __shared__ float hA[16 * 64];
  __shared__ float hB[16 * 64];
  __shared__ float tpw[16 * 512];
  __shared__ int   sidx[16];
  __shared__ int   ridx[16];
  __shared__ float ya[16 * 4];

  const int e0 = blockIdx.x * 16;
  const int lane = threadIdx.x;
  const int col = lane & 15, g = lane >> 4;

  if (lane < 16) {
    sidx[lane] = edge_index[e0 + lane];
    ridx[lane] = edge_index[N_EDGES + e0 + lane];
  }
  for (int i = lane; i < 16 * 8; i += 32) ef[i] = edge_feats[(size_t)e0 * 8 + i];
  for (int i = lane; i < 16 * 4; i += 32) ya[i] = edge_attrs[(size_t)e0 * 4 + i];
  for (int i = lane; i < 8 * 64; i += 32) wfc1[i] = W_fc1[i];
  __syncthreads();

  // Prefetch the gathered sender rows so the random-access latency overlaps
  // the MLP below (x0u row = 512B -> 4 lines, x1u row = 1536B -> 12 lines).
  {
    const int e = lane >> 1;  // 2 lanes per edge
    const int half = lane & 1;
    const char* p0 = (const char*)(x0u + (size_t)sidx[e] * 128);
    const char* p1 = (const char*)(x1u + (size_t)sidx[e] * 384);
#pragma unroll
    for (int l = half; l < 4; l += 2) __builtin_prefetch(p0 + l * 128, 0, 0);
#pragma unroll
    for (int l = half; l < 12; l += 2) __builtin_prefetch(p1 + l * 128, 0, 0);
  }

  // Layer 1 (K=8, VALU): hA = silu(ef @ W_fc1 / sqrt(8))
  for (int i = lane; i < 16 * 64; i += 32) {
    int r = i >> 6, c = i & 63;
    float acc = 0.0f;
#pragma unroll
    for (int k = 0; k < 8; ++k) acc += ef[r * 8 + k] * wfc1[k * 64 + c];
    hA[i] = silu(acc * 0.35355339059327373f);
  }
  __syncthreads();

  // Layer 2: hB = silu(hA @ Wfc2 / 8)   (K=64, N=64)
  {
    v16bf a[2];
#pragma unroll
    for (int kt = 0; kt < 2; ++kt) a[kt] = load_a_frag(hA + kt * 32, 64, 1);
    BFrags<2> cb, nb;
    load_bfrags<2>(cb, Wfc2, 0, 4);
#pragma unroll
    for (int nt = 0; nt < 4; ++nt) {
      if (nt < 3) load_bfrags<2>(nb, Wfc2, nt + 1, 4);
      v8f c = wmma_chain<2>(a, cb);
#pragma unroll
      for (int r = 0; r < 8; ++r)
        hB[(r + 8 * g) * 64 + nt * 16 + col] = silu(c[r] * 0.125f);
      cb = nb;
    }
  }
  __syncthreads();

  // Layer 3: hA = silu(hB @ Wfc3 / 8)
  {
    v16bf a[2];
#pragma unroll
    for (int kt = 0; kt < 2; ++kt) a[kt] = load_a_frag(hB + kt * 32, 64, 1);
    BFrags<2> cb, nb;
    load_bfrags<2>(cb, Wfc3, 0, 4);
#pragma unroll
    for (int nt = 0; nt < 4; ++nt) {
      if (nt < 3) load_bfrags<2>(nb, Wfc3, nt + 1, 4);
      v8f c = wmma_chain<2>(a, cb);
#pragma unroll
      for (int r = 0; r < 8; ++r)
        hA[(r + 8 * g) * 64 + nt * 16 + col] = silu(c[r] * 0.125f);
      cb = nb;
    }
  }
  __syncthreads();

  // Layer 4: tpw = hA @ Wfc4 / 8   (K=64, N=512), double-buffered B
  {
    v16bf a[2];
#pragma unroll
    for (int kt = 0; kt < 2; ++kt) a[kt] = load_a_frag(hA + kt * 32, 64, 1);
    BFrags<2> cb, nb;
    load_bfrags<2>(cb, Wfc4, 0, 32);
#pragma unroll
    for (int nt = 0; nt < 32; ++nt) {
      if (nt < 31) load_bfrags<2>(nb, Wfc4, nt + 1, 32);
      v8f c = wmma_chain<2>(a, cb);
#pragma unroll
      for (int r = 0; r < 8; ++r)
        tpw[(r + 8 * g) * 512 + nt * 16 + col] = c[r] * 0.125f;
      cb = nb;
    }
  }
  __syncthreads();

  // Tensor-product messages + scatter-add
  const float inv_sqrt3 = 0.5773502691896258f;
  for (int e = 0; e < 16; ++e) {
    const int s  = sidx[e];
    const int rr = ridx[e];
    const float y0  = ya[e * 4 + 0];
    const float y1a = ya[e * 4 + 1];
    const float y1b = ya[e * 4 + 2];
    const float y1c = ya[e * 4 + 3];
    const float* w    = tpw + e * 512;
    const float* xs0p = x0u + (size_t)s * 128;
    const float* xs1p = x1u + (size_t)s * 384;
    float* m0r = m0 + (size_t)rr * 256;
    float* m1r = m1 + (size_t)rr * 768;
#pragma unroll
    for (int ub = 0; ub < 4; ++ub) {
      const int u = ub * 32 + lane;
      const float xs0v = xs0p[u];
      const float w0 = w[u], w1 = w[128 + u], w2 = w[256 + u], w3 = w[384 + u];
      const float xa = xs1p[u * 3 + 0];
      const float xb = xs1p[u * 3 + 1];
      const float xc = xs1p[u * 3 + 2];
      atomicAdd(&m0r[u], xs0v * y0 * w0);                                   // p0
      atomicAdd(&m0r[128 + u],
                (xa * y1a + xb * y1b + xc * y1c) * w3 * inv_sqrt3);         // p3
      const float q = xs0v * w1;                                            // p1
      atomicAdd(&m1r[u * 3 + 0], q * y1a);
      atomicAdd(&m1r[u * 3 + 1], q * y1b);
      atomicAdd(&m1r[u * 3 + 2], q * y1c);
      const float yw2 = y0 * w2;                                            // p2
      atomicAdd(&m1r[384 + u * 3 + 0], xa * yw2);
      atomicAdd(&m1r[384 + u * 3 + 1], xb * yw2);
      atomicAdd(&m1r[384 + u * 3 + 2], xc * yw2);
    }
  }
}

// ---------------------------------------------------------------------------
// Output: o0/o1 = linear-down (K=256), then element-dependent skip
// s[n,k] = sum_v attrs[n,v] * (o @ W_sk[:,v,:])[n,k] * invs.
// Skip phase holds all 16 A fragments (o0 + 3x o1); per (nt,v) one Wsk0 and
// one Wsk1 fragment set feed FOUR wmma chains (B reuse across outputs).
// ---------------------------------------------------------------------------
__global__ __launch_bounds__(32) void k_node_out(
    const float* __restrict__ node_attrs,
    const float* __restrict__ m0, const float* __restrict__ m1,
    const bf16_t* __restrict__ Wlin0, const bf16_t* __restrict__ Wlin1,
    const bf16_t* __restrict__ Wsk0, const bf16_t* __restrict__ Wsk1,
    float* __restrict__ out) {
  __shared__ float stage[16 * 256];
  __shared__ float o0[16 * 128];
  __shared__ float o1[16 * 384];
  __shared__ float attrs[16 * N_ELEM];
  const int n0 = blockIdx.x * 16;
  const int lane = threadIdx.x;
  const int col = lane & 15, g = lane >> 4;
  const float inv2 = 0.00390625f;            // 1/(sqrt(256)*16)
  const float invs = 0.027950849718747374f;  // 1/sqrt(1280)

  for (int i = lane; i < 16 * N_ELEM; i += 32)
    attrs[i] = node_attrs[(size_t)n0 * N_ELEM + i];

  // o0 = m0 @ Wlin0 * inv2 : async-stage 16x256 tile
  {
    const float* gsrc = m0 + (size_t)n0 * 256;
    for (int i = lane * 4; i < 16 * 256; i += 32 * 4)
      async_copy_b128(&stage[i], gsrc + i);
    wait_async0();
  }
  __syncthreads();
  {
    v16bf a[8];
#pragma unroll
    for (int kt = 0; kt < 8; ++kt) a[kt] = load_a_frag(stage + kt * 32, 256, 1);
#pragma unroll
    for (int nt = 0; nt < 8; ++nt) {
      BFrags<8> bb;
      load_bfrags<8>(bb, Wlin0, nt, 8);
      v8f c = wmma_chain<8>(a, bb);
#pragma unroll
      for (int r = 0; r < 8; ++r)
        o0[(r + 8 * g) * 128 + nt * 16 + col] = c[r] * inv2;
    }
  }
  __syncthreads();

  // o1[:, :, mc] = m1[:, :, mc] @ Wlin1 * inv2
  for (int mc = 0; mc < 3; ++mc) {
    for (int i = lane; i < 16 * 256; i += 32) {
      int r = i >> 8, u = i & 255;
      stage[i] = m1[((size_t)(n0 + r) * 256 + u) * 3 + mc];
    }
    __syncthreads();
    {
      v16bf a[8];
#pragma unroll
      for (int kt = 0; kt < 8; ++kt)
        a[kt] = load_a_frag(stage + kt * 32, 256, 1);
#pragma unroll
      for (int nt = 0; nt < 8; ++nt) {
        BFrags<8> bb;
        load_bfrags<8>(bb, Wlin1, nt, 8);
        v8f c = wmma_chain<8>(a, bb);
#pragma unroll
        for (int r = 0; r < 8; ++r)
          o1[(r + 8 * g) * 384 + (nt * 16 + col) * 3 + mc] = c[r] * inv2;
      }
    }
    __syncthreads();
  }

  // Skip contraction: all 16 A fragments live in VGPRs; per (nt,v) one Wsk0
  // and one Wsk1 fragment set drive 4 wmma chains (s0, s1x, s1y, s1z).
  {
    v16bf a0[4], a1[3][4];
#pragma unroll
    for (int kt = 0; kt < 4; ++kt) {
      a0[kt] = load_a_frag(o0 + kt * 32, 128, 1);
#pragma unroll
      for (int mc = 0; mc < 3; ++mc)
        a1[mc][kt] = load_a_frag(o1 + mc + kt * 96, 384, 3);
    }
    for (int nt = 0; nt < 8; ++nt) {
      float acc[4][8];
#pragma unroll
      for (int q = 0; q < 4; ++q)
#pragma unroll
        for (int r = 0; r < 8; ++r) acc[q][r] = 0.0f;

      for (int v = 0; v < N_ELEM; ++v) {
        BFrags<4> b0, b1;
        load_bfrags<4>(b0, Wsk0 + (size_t)v * 16384, nt, 8);
        load_bfrags<4>(b1, Wsk1 + (size_t)v * 16384, nt, 8);
        float av[8];
#pragma unroll
        for (int r = 0; r < 8; ++r) av[r] = attrs[(r + 8 * g) * N_ELEM + v];
        v8f c0 = wmma_chain<4>(a0, b0);
        v8f c1 = wmma_chain<4>(a1[0], b1);
        v8f c2 = wmma_chain<4>(a1[1], b1);
        v8f c3 = wmma_chain<4>(a1[2], b1);
#pragma unroll
        for (int r = 0; r < 8; ++r) {
          acc[0][r] += av[r] * c0[r];
          acc[1][r] += av[r] * c1[r];
          acc[2][r] += av[r] * c2[r];
          acc[3][r] += av[r] * c3[r];
        }
      }
#pragma unroll
      for (int r = 0; r < 8; ++r) {
        const size_t base = ((size_t)(n0 + r + 8 * g) * 128 + nt * 16 + col) * 4;
        out[base + 0] = acc[0][r] * invs;
        out[base + 1] = acc[1][r] * invs;
        out[base + 2] = acc[2][r] * invs;
        out[base + 3] = acc[3][r] * invs;
      }
    }
  }
}

// ---------------------------------------------------------------------------
extern "C" void kernel_launch(void* const* d_in, const int* in_sizes, int n_in,
                              void* d_out, int out_size, void* d_ws,
                              size_t ws_size, hipStream_t stream) {
  (void)in_sizes; (void)n_in; (void)out_size; (void)ws_size;

  const float* node_attrs = (const float*)d_in[0];
  const float* node_feats = (const float*)d_in[1];
  const float* edge_attrs = (const float*)d_in[2];
  const float* edge_feats = (const float*)d_in[3];
  const int*   edge_index = (const int*)d_in[4];
  const float* W_up0 = (const float*)d_in[5];
  const float* W_up1 = (const float*)d_in[6];
  const float* W_fc1 = (const float*)d_in[7];
  const float* W_fc2 = (const float*)d_in[8];
  const float* W_fc3 = (const float*)d_in[9];
  const float* W_fc4 = (const float*)d_in[10];
  const float* W_lin0 = (const float*)d_in[11];
  const float* W_lin1 = (const float*)d_in[12];
  const float* W_sk0 = (const float*)d_in[13];
  const float* W_sk1 = (const float*)d_in[14];
  float* out = (float*)d_out;

  // Workspace layout (~124 MB): packed bf16 weights | x0u | x1u | m0 | m1
  uint8_t* ws = (uint8_t*)d_ws;
  bf16_t* wup0  = (bf16_t*)ws;
  bf16_t* wup1  = wup0 + 16384;
  bf16_t* wfc2  = wup1 + 16384;
  bf16_t* wfc3  = wfc2 + 4096;
  bf16_t* wfc4  = wfc3 + 4096;
  bf16_t* wlin0 = wfc4 + 32768;
  bf16_t* wlin1 = wlin0 + 32768;
  bf16_t* wsk0  = wlin1 + 32768;
  bf16_t* wsk1  = wsk0 + 163840;
  float* fbase = (float*)(ws + 933888);
  float* x0u = fbase;                        // 20000*128
  float* x1u = x0u + (size_t)N_NODES * 128;  // 20000*384
  float* m0  = x1u + (size_t)N_NODES * 384;  // 20000*256
  float* m1  = m0 + (size_t)N_NODES * 256;   // 20000*768

  // 1) pre-swizzle weights into WMMA B-fragment order (bf16)
  auto pk = [&](const float* s, bf16_t* d, int K, int N, int ld) {
    int n = K * N;
    k_pack_b<<<(n + 255) / 256, 256, 0, stream>>>(s, d, K, N, ld);
  };
  pk(W_up0, wup0, 128, 128, 128);
  pk(W_up1, wup1, 128, 128, 128);
  pk(W_fc2, wfc2, 64, 64, 64);
  pk(W_fc3, wfc3, 64, 64, 64);
  pk(W_fc4, wfc4, 64, 512, 512);
  pk(W_lin0, wlin0, 256, 128, 128);
  pk(W_lin1, wlin1, 256, 128, 128);
  k_pack_sk<<<(N_ELEM * 16384 + 255) / 256, 256, 0, stream>>>(W_sk0, wsk0);
  k_pack_sk<<<(N_ELEM * 16384 + 255) / 256, 256, 0, stream>>>(W_sk1, wsk1);

  // 2) zero scatter accumulators (m0 and m1 are contiguous)
  {
    int n = N_NODES * (256 + 768);
    k_zero<<<(n + 255) / 256, 256, 0, stream>>>(m0, n);
  }

  // 3) node up-projection (WMMA, async LDS staging)
  k_node_up<<<N_NODES / 16, 32, 0, stream>>>(node_feats, wup0, wup1, x0u, x1u);

  // 4) fused edge MLP + tensor-product messages + scatter (WMMA + prefetch + atomics)
  k_edge<<<N_EDGES / 16, 32, 0, stream>>>(edge_attrs, edge_feats, edge_index,
                                          W_fc1, wfc2, wfc3, wfc4, x0u, x1u,
                                          m0, m1);

  // 5) linear-down + element-dependent skip contraction (WMMA)
  k_node_out<<<N_NODES / 16, 32, 0, stream>>>(node_attrs, m0, m1, wlin0, wlin1,
                                              wsk0, wsk1, out);
}
